// InvariantPointAttention_90855738179781
// MI455X (gfx1250) — matrix operand order repaired and checked by reading, not compiled, not run on hardware
//
#include <hip/hip_runtime.h>
#include <hip/hip_bf16.h>
#include <math.h>

// ---------------------------------------------------------------------------
// Invariant Point Attention for MI455X (gfx1250).
// bf16 WMMA GEMMs + fused per-row attention with async global->LDS streaming
// of the pair tensor (the bandwidth-dominant operand: 134 MB, HBM floor ~6us).
// Shapes: B=1, N=512, CS=384, CZ=128, H=12, C=16, PQ=4, PV=8
// ---------------------------------------------------------------------------

#define NSEQ 512
#define CS_  384
#define CZ_  128
#define HH   12
#define CC   16
#define PQ_  4
#define PV_  8

#define PROJ_COLS 1152          // 192(q)+192(k)+192(v)+144(qf)+144(kf)+288(vf)
#define CAT_COLS  2016          // 192(o_rep)+288(o_pt_local)+1536(o_pair)

#define JC    16                // pair rows per async chunk (8 KB / chunk)
#define NCHUNK (NSEQ / JC)      // 32 chunks

#define W_C_CONST 0.23570226039551584f   // sqrt(2/(9*PQ)) = sqrt(1/18)
#define W_L_CONST 0.5773502691896258f    // sqrt(1/3)

typedef __attribute__((ext_vector_type(16))) __bf16 v16bf;
typedef __attribute__((ext_vector_type(8)))  float  v8f;

// ---------------- WMMA tile loaders (ISA 7.12.2 layouts, wave32) -----------
// A: 16x32 bf16, row-major src.  lane<16 holds K{0..7,16..23}, lane>=16 holds
// K{8..15,24..31} of row M=lane&15.
__device__ __forceinline__ v16bf load_a_bf16(const __bf16* A, int lda) {
  const int lane = threadIdx.x & 31;
  const int row  = lane & 15;
  const int khalf = (lane >> 4) << 3;      // 0 or 8
  v16bf a;
#pragma unroll
  for (int i = 0; i < 16; ++i) {
    int k = ((i & 8) << 1) + khalf + (i & 7);   // i<8 -> K base 0, i>=8 -> 16
    a[i] = A[row * lda + k];
  }
  return a;
}

// B: 32x16 bf16, row-major (K x N).  lanes 0-15: K{0..7,16..23}, lanes 16-31: K{8..15,24..31}
__device__ __forceinline__ v16bf load_b_bf16(const __bf16* B, int ldb) {
  const int lane = threadIdx.x & 31;
  const int n  = lane & 15;
  const int kb = (lane >> 4) << 3;
  v16bf b;
#pragma unroll
  for (int i = 0; i < 16; ++i) b[i] = B[(kb + ((i & 8) << 1) + (i & 7)) * ldb + n];
  return b;
}

// ---------------- async global->LDS copy of one 8 KB pair chunk -------------
// 256 threads x 2 x 16B = 8 KB (JC=16 rows of 128 fp32). Tracked by ASYNCcnt.
__device__ __forceinline__ void async_copy_chunk(const float* gbase,
                                                 unsigned goff, unsigned ldsoff,
                                                 int tid) {
#pragma unroll
  for (int r = 0; r < 2; ++r) {
    const unsigned seg = (unsigned)(r * 256 + tid);
    const unsigned l = ldsoff + seg * 16u;
    const unsigned g = goff  + seg * 16u;
    asm volatile("global_load_async_to_lds_b128 %0, %1, %2"
                 :: "v"(l), "v"(g), "s"(gbase) : "memory");
  }
}

#define ASYNC_WAIT_2() asm volatile("s_wait_asynccnt 0x2" ::: "memory")
#define ASYNC_WAIT_0() asm volatile("s_wait_asynccnt 0x0" ::: "memory")

// ---------------- K0: fp32 -> bf16 repack of GEMM operands ------------------
__global__ __launch_bounds__(256)
void ipa_repack_kernel(const float* __restrict__ s,
                       const float* __restrict__ Wq,  const float* __restrict__ Wk,
                       const float* __restrict__ Wv,  const float* __restrict__ Wqf,
                       const float* __restrict__ Wkf, const float* __restrict__ Wvf,
                       const float* __restrict__ Wo,
                       __bf16* __restrict__ sbf, __bf16* __restrict__ wcat,
                       __bf16* __restrict__ wobf) {
  int t = blockIdx.x * blockDim.x + threadIdx.x;
  const int NS = NSEQ * CS_;
  const int NW = CS_ * PROJ_COLS;
  const int NO = CAT_COLS * CS_;
  if (t < NS) { sbf[t] = (__bf16)s[t]; return; }
  t -= NS;
  if (t < NW) {
    int r = t / PROJ_COLS, c = t - r * PROJ_COLS;
    float v;
    if      (c < 192) v = Wq [r * 192 + c];
    else if (c < 384) v = Wk [r * 192 + (c - 192)];
    else if (c < 576) v = Wv [r * 192 + (c - 384)];
    else if (c < 720) v = Wqf[r * 144 + (c - 576)];
    else if (c < 864) v = Wkf[r * 144 + (c - 720)];
    else              v = Wvf[r * 288 + (c - 864)];
    wcat[t] = (__bf16)v; return;
  }
  t -= NW;
  if (t < NO) wobf[t] = (__bf16)Wo[t];
}

// ---------------- K1/K4: bf16 WMMA GEMM, one wave per 16x16 C tile ----------
__global__ __launch_bounds__(32)
void ipa_gemm_bf16_kernel(const __bf16* __restrict__ A, int lda,
                          const __bf16* __restrict__ B, int ldb,
                          float* __restrict__ C, int ldc, int K,
                          const float* __restrict__ bias) {
  const int tn = blockIdx.x, tm = blockIdx.y;
  v8f acc = {};
  const __bf16* Abase = A + (size_t)tm * 16 * lda;
  const __bf16* Bbase = B + tn * 16;
  for (int k = 0; k < K; k += 32) {
    v16bf a = load_a_bf16(Abase + k, lda);
    v16bf b = load_b_bf16(Bbase + (size_t)k * ldb, ldb);
    acc = __builtin_amdgcn_wmma_f32_16x16x32_bf16(false, a, false, b,
                                                  (short)0, acc, false, false);
  }
  const int lane = threadIdx.x & 31;
  const int n  = lane & 15;
  const int mb = (lane >> 4) << 3;
  const float bv = bias ? bias[tn * 16 + n] : 0.0f;
#pragma unroll
  for (int r = 0; r < 8; ++r)
    C[(size_t)(tm * 16 + mb + r) * ldc + tn * 16 + n] = acc[r] + bv;
}

// ---------------- K2: build global frames gq/gk/gv and norms sq/sk ----------
__global__ __launch_bounds__(256)
void ipa_frames_kernel(const float* __restrict__ P,
                       const float* __restrict__ rot, const float* __restrict__ trans,
                       float* __restrict__ gq, float* __restrict__ gk,
                       float* __restrict__ gv,
                       float* __restrict__ sq, float* __restrict__ sk) {
  int t = blockIdx.x * blockDim.x + threadIdx.x;
  if (t >= NSEQ * HH) return;
  const int i = t / HH, h = t - i * HH;
  float R[9], tv[3];
#pragma unroll
  for (int u = 0; u < 9; ++u) R[u] = rot[i * 9 + u];
#pragma unroll
  for (int u = 0; u < 3; ++u) tv[u] = trans[i * 3 + u];
  const float* Pr = P + (size_t)i * PROJ_COLS;
  float s1 = 0.f, s2 = 0.f;
#pragma unroll
  for (int q = 0; q < PQ_; ++q) {
    float f[3], g[3];
#pragma unroll
    for (int p = 0; p < 3; ++p) f[p] = Pr[576 + h * 12 + q * 3 + p];   // qf
#pragma unroll
    for (int o = 0; o < 3; ++o) {
      g[o] = R[o*3+0]*f[0] + R[o*3+1]*f[1] + R[o*3+2]*f[2] + tv[o];
      gq[(size_t)i * 144 + h * 12 + q * 3 + o] = g[o];
      s1 += g[o] * g[o];
    }
#pragma unroll
    for (int p = 0; p < 3; ++p) f[p] = Pr[720 + h * 12 + q * 3 + p];   // kf
#pragma unroll
    for (int o = 0; o < 3; ++o) {
      g[o] = R[o*3+0]*f[0] + R[o*3+1]*f[1] + R[o*3+2]*f[2] + tv[o];
      gk[(size_t)i * 144 + h * 12 + q * 3 + o] = g[o];
      s2 += g[o] * g[o];
    }
  }
  sq[i * HH + h] = s1;
  sk[i * HH + h] = s2;
#pragma unroll
  for (int q = 0; q < PV_; ++q) {                                       // vf
    float f[3];
#pragma unroll
    for (int p = 0; p < 3; ++p) f[p] = Pr[864 + h * 24 + q * 3 + p];
#pragma unroll
    for (int o = 0; o < 3; ++o)
      gv[(size_t)i * 288 + h * 24 + q * 3 + o] =
          R[o*3+0]*f[0] + R[o*3+1]*f[1] + R[o*3+2]*f[2] + tv[o];
  }
}

// ---------------- K3: fused attention, one workgroup per query row i --------
// LDS pool (manually overlaid phases, ~59 KB total):
//   A [0,24576)      : fp32 scores (pass1+softmax)  | bf16 B-tile 32x128 (WMMA)
//   B [24576,40960)  : 2x8KB async pair staging     | softmax reduction scratch
//   C [40960,57344)  : bf16 attn^T 16x512 WMMA A-tile (softmax .. end)
__global__ __launch_bounds__(256)
void ipa_attn_kernel(const float* __restrict__ P,
                     const float* __restrict__ pair,
                     const float* __restrict__ Wb,
                     const float* __restrict__ gq, const float* __restrict__ gk,
                     const float* __restrict__ gv,
                     const float* __restrict__ sq, const float* __restrict__ sk,
                     const float* __restrict__ rot, const float* __restrict__ trans,
                     const float* __restrict__ head_w,
                     __bf16* __restrict__ catbf) {
  const int i   = blockIdx.x;
  const int tid = threadIdx.x;

  __shared__ __align__(16) char pool[24576 + 16384 + 16384];
  float*  sc    = (float*)pool;                  // 512*12 fp32 scores
  __bf16* btile = (__bf16*)pool;                 // 32x128 bf16 (phase 3)
  float*  stage = (float*)(pool + 24576);        // 2 x (16x128) fp32 staging
  float*  red   = (float*)(pool + 24576);        // 256*12 fp32 (softmax)
  __bf16* abf   = (__bf16*)(pool + 40960);       // 16x512 bf16 attn^T

  __shared__ float qi[HH * CC], gqi[HH * 12], sqi[HH];
  __shared__ float Ri[9], ti[3], coef[HH], hmax[HH], hsum[HH];
  __shared__ float optv[HH * PV_ * 3];

  const unsigned stage_lds = (unsigned)(uintptr_t)(pool + 24576);
  const unsigned gbase0    = (unsigned)((size_t)i * NSEQ * CZ_ * 4u);

  // ---- stage per-row constants ----
  if (tid < HH * CC) qi[tid]  = P[(size_t)i * PROJ_COLS + tid];       // q block
  if (tid < HH * 12) gqi[tid] = gq[(size_t)i * 144 + tid];
  if (tid < HH) {
    sqi[tid] = sq[i * HH + tid];
    float x = head_w[tid];
    float g = (x > 20.f) ? x : log1pf(expf(x));                        // softplus
    float g2 = g*g, g4 = g2*g2, g12 = g4*g4*g4;                        // gamma^H
    coef[tid] = -0.5f * g12 * W_C_CONST;
  }
  if (tid < 9) Ri[tid] = rot[i * 9 + tid];
  if (tid < 3) ti[tid] = trans[i * 3 + tid];
  __syncthreads();

  // =========== pass 1: scores, pair streamed via async-to-LDS ==============
  float lmax[HH];
#pragma unroll
  for (int h = 0; h < HH; ++h) lmax[h] = -1e30f;

  async_copy_chunk(pair, gbase0, stage_lds, tid);            // prime chunk 0
  for (int ch = 0; ch < NCHUNK; ++ch) {
    const int buf = ch & 1;
    if (ch + 1 < NCHUNK) {
      async_copy_chunk(pair, gbase0 + (unsigned)(ch + 1) * (JC * CZ_ * 4),
                       stage_lds + (unsigned)((ch + 1) & 1) * (JC * CZ_ * 4), tid);
      ASYNC_WAIT_2();                      // chunk ch landed; ch+1 in flight
    } else {
      ASYNC_WAIT_0();
    }
    __syncthreads();

    if (tid < JC * HH) {                                   // 192 score entries
      const int jr = tid / HH, h = tid - jr * HH;
      const int j  = ch * JC + jr;
      const float* Pj = P + (size_t)j * PROJ_COLS;
      float dqk = 0.f;
#pragma unroll
      for (int c = 0; c < CC; ++c) dqk += qi[h * CC + c] * Pj[192 + h * CC + c];
      float bia = 0.f;
      const float* pr = stage + buf * (JC * CZ_) + jr * CZ_;
#pragma unroll 8
      for (int z = 0; z < CZ_; ++z) bia += pr[z] * Wb[z * HH + h];
      float qkpt = 0.f;
      const float* gkj = gk + (size_t)j * 144 + h * 12;
#pragma unroll
      for (int u = 0; u < 12; ++u) qkpt += gqi[h * 12 + u] * gkj[u];
      const float d2 = sqi[h] + sk[j * HH + h] - 2.f * qkpt;
      const float sv = W_L_CONST * (dqk * (1.f / 16.f) + bia + coef[h] * d2);
      sc[j * HH + h] = sv;
      lmax[h] = fmaxf(lmax[h], sv);
    }
    __syncthreads();                       // staging buffer reuse fence
  }

  // =========== softmax over j per head (red overlays dead staging) =========
#pragma unroll
  for (int h = 0; h < HH; ++h) red[tid * HH + h] = lmax[h];
  __syncthreads();
  for (int off = 128; off > 0; off >>= 1) {
    if (tid < off)
#pragma unroll
      for (int h = 0; h < HH; ++h)
        red[tid * HH + h] = fmaxf(red[tid * HH + h], red[(tid + off) * HH + h]);
    __syncthreads();
  }
  if (tid < HH) hmax[tid] = red[tid];
  __syncthreads();

  float lsum[HH];
#pragma unroll
  for (int h = 0; h < HH; ++h) lsum[h] = 0.f;
  for (int e = tid; e < NSEQ * HH; e += 256) {
    const int h = e % HH;
    const float ex = expf(sc[e] - hmax[h]);
    sc[e] = ex;
    lsum[h] += ex;
  }
  __syncthreads();
#pragma unroll
  for (int h = 0; h < HH; ++h) red[tid * HH + h] = lsum[h];
  __syncthreads();
  for (int off = 128; off > 0; off >>= 1) {
    if (tid < off)
#pragma unroll
      for (int h = 0; h < HH; ++h)
        red[tid * HH + h] += red[(tid + off) * HH + h];
    __syncthreads();
  }
  if (tid < HH) hsum[tid] = red[tid];
  __syncthreads();

  for (int e = tid; e < NSEQ * HH; e += 256) {
    const int j = e / HH, h = e - j * HH;
    abf[h * NSEQ + j] = (__bf16)(sc[e] / hsum[h]);   // attn^T bf16 A-tile
  }
  for (int e = tid; e < 4 * NSEQ; e += 256) abf[HH * NSEQ + e] = (__bf16)0.f;
  __syncthreads();                                   // sc dead from here

  // =========== o_pair = attn^T(16x512) @ pair[i](512x128), WMMA ============
  // Stream pair again (L2-resident: 134 MB < 192 MB L2) via async-to-LDS,
  // convert each 32x128 K-panel to bf16 in LDS, 8 waves x 1 z-tile each.
  {
    const int wave = tid >> 5;
    v8f acc = {};
    async_copy_chunk(pair, gbase0, stage_lds, tid);          // prime chunk 0
    for (int ch = 0; ch < NCHUNK; ++ch) {
      const int buf = ch & 1;
      if (ch + 1 < NCHUNK) {
        async_copy_chunk(pair, gbase0 + (unsigned)(ch + 1) * (JC * CZ_ * 4),
                         stage_lds + (unsigned)((ch + 1) & 1) * (JC * CZ_ * 4), tid);
        ASYNC_WAIT_2();
      } else {
        ASYNC_WAIT_0();
      }
      __syncthreads();
      // convert this 16-row chunk into its half of the 32x128 bf16 B panel
      for (int e = tid; e < JC * CZ_; e += 256)
        btile[(buf * JC + (e >> 7)) * CZ_ + (e & 127)] =
            (__bf16)stage[buf * (JC * CZ_) + e];
      __syncthreads();
      if (buf == 1) {                        // full 32-row K panel ready
        const int kc = (ch - 1) * JC;        // K offset into attn^T
        v16bf a = load_a_bf16((const __bf16*)abf + kc, NSEQ);
        v16bf b = load_b_bf16(btile + wave * 16, CZ_);
        acc = __builtin_amdgcn_wmma_f32_16x16x32_bf16(false, a, false, b,
                                                      (short)0, acc, false, false);
      }
      __syncthreads();                       // panel/staging reuse fence
    }
    const int lane = tid & 31;
    const int n  = lane & 15;
    const int mb = (lane >> 4) << 3;
#pragma unroll
    for (int r = 0; r < 8; ++r) {
      const int h = mb + r;
      if (h < HH)
        catbf[(size_t)i * CAT_COLS + 480 + h * CZ_ + wave * 16 + n] = (__bf16)acc[r];
    }
  }

  // =========== o_rep = attn @ v (attn read from bf16 LDS tile) =============
  if (tid < HH * CC) {
    const int h = tid >> 4, d = tid & 15;
    float acc = 0.f;
    for (int j = 0; j < NSEQ; ++j) {
      if ((j & 7) == 0 && j + 8 < NSEQ)
        __builtin_prefetch(P + (size_t)(j + 8) * PROJ_COLS + 384 + h * CC + d, 0, 1);
      acc += (float)abf[h * NSEQ + j] * P[(size_t)j * PROJ_COLS + 384 + h * CC + d];
    }
    catbf[(size_t)i * CAT_COLS + h * CC + d] = (__bf16)acc;
  }

  // =========== o_pt = attn @ gv, then frame-inverse rotate =================
  if (tid < HH * PV_ * 3) {
    const int h = tid / 24, rem = tid % 24;
    float acc = 0.f;
    for (int j = 0; j < NSEQ; ++j)
      acc += (float)abf[h * NSEQ + j] * gv[(size_t)j * 288 + h * 24 + rem];
    optv[tid] = acc;
  }
  __syncthreads();
  if (tid < HH * PV_ * 3) {
    const int h = tid / 24, rem = tid % 24, q = rem / 3, p = rem % 3;
    float r = 0.f;
#pragma unroll
    for (int o = 0; o < 3; ++o)
      r += Ri[o * 3 + p] * (optv[h * 24 + q * 3 + o] - ti[o]);
    catbf[(size_t)i * CAT_COLS + 192 + h * 24 + q * 3 + p] = (__bf16)r;
  }
}

// ---------------------------------------------------------------------------
extern "C" void kernel_launch(void* const* d_in, const int* in_sizes, int n_in,
                              void* d_out, int out_size, void* d_ws, size_t ws_size,
                              hipStream_t stream) {
  const float* s      = (const float*)d_in[0];
  const float* pair   = (const float*)d_in[1];
  const float* rot    = (const float*)d_in[2];
  const float* trans  = (const float*)d_in[3];
  const float* Wq     = (const float*)d_in[4];
  const float* Wk     = (const float*)d_in[5];
  const float* Wv     = (const float*)d_in[6];
  const float* Wqf    = (const float*)d_in[7];
  const float* Wkf    = (const float*)d_in[8];
  const float* Wvf    = (const float*)d_in[9];
  const float* Wb     = (const float*)d_in[10];
  const float* Wo     = (const float*)d_in[11];
  const float* bo     = (const float*)d_in[12];
  const float* head_w = (const float*)d_in[13];
  float* out = (float*)d_out;

  // -------- workspace layout (≈ 8.1 MB, all offsets 256B-aligned) ----------
  char* ws = (char*)d_ws;
  size_t off = 0;
  __bf16* sbf  = (__bf16*)(ws + off); off += (size_t)NSEQ * CS_ * 2;
  __bf16* wcat = (__bf16*)(ws + off); off += (size_t)CS_ * PROJ_COLS * 2;
  __bf16* wobf = (__bf16*)(ws + off); off += (size_t)CAT_COLS * CS_ * 2;
  float*  P    = (float*) (ws + off); off += (size_t)NSEQ * PROJ_COLS * 4;
  float*  gq   = (float*) (ws + off); off += (size_t)NSEQ * 144 * 4;
  float*  gk   = (float*) (ws + off); off += (size_t)NSEQ * 144 * 4;
  float*  gv   = (float*) (ws + off); off += (size_t)NSEQ * 288 * 4;
  float*  sq   = (float*) (ws + off); off += (size_t)NSEQ * HH * 4;
  float*  sk   = (float*) (ws + off); off += (size_t)NSEQ * HH * 4;
  __bf16* catb = (__bf16*)(ws + off); off += (size_t)NSEQ * CAT_COLS * 2;

  // K0: repack operands to bf16
  {
    const int total = NSEQ * CS_ + CS_ * PROJ_COLS + CAT_COLS * CS_;
    ipa_repack_kernel<<<(total + 255) / 256, 256, 0, stream>>>(
        s, Wq, Wk, Wv, Wqf, Wkf, Wvf, Wo, sbf, wcat, wobf);
  }

  // K1: P = s @ [Wq|Wk|Wv|Wqf|Wkf|Wvf]   (512x384 @ 384x1152, bf16 WMMA)
  ipa_gemm_bf16_kernel<<<dim3(PROJ_COLS / 16, NSEQ / 16), 32, 0, stream>>>(
      sbf, CS_, wcat, PROJ_COLS, P, PROJ_COLS, CS_, nullptr);

  // K2: global frames + norms
  ipa_frames_kernel<<<(NSEQ * HH + 255) / 256, 256, 0, stream>>>(
      P, rot, trans, gq, gk, gv, sq, sk);

  // K3: fused attention (async pair streaming + softmax + o_rep/o_pt/o_pair)
  ipa_attn_kernel<<<NSEQ, 256, 0, stream>>>(
      P, pair, Wb, gq, gk, gv, sq, sk, rot, trans, head_w, catb);

  // K4: out = cat @ Wo + bo   (512x2016 @ 2016x384, bf16 WMMA)
  ipa_gemm_bf16_kernel<<<dim3(CS_ / 16, NSEQ / 16), 32, 0, stream>>>(
      catb, CAT_COLS, wobf, CS_, out, CS_, CAT_COLS, bo);
}
